// SparsePatchWordAlignment_57724360458465
// MI455X (gfx1250) — matrix operand order repaired and checked by reading, not compiled, not run on hardware
//
#include <hip/hip_runtime.h>

typedef __attribute__((ext_vector_type(16))) __bf16          v16bf;
typedef __attribute__((ext_vector_type(2)))  __bf16          v2bf;
typedef __attribute__((ext_vector_type(8)))  float           v8f;
typedef __attribute__((ext_vector_type(2)))  float           f32x2;
typedef __attribute__((ext_vector_type(4)))  float           f32x4;
typedef __attribute__((ext_vector_type(8)))  unsigned short  v8u;
typedef __attribute__((ext_vector_type(4)))  unsigned short  v4u;

#define B_BATCH   64
#define P_DIM     4096
#define W_DIM     256
#define D_DIM     512
#define LDS_PITCH 520                     // halfwords per text row (512 + 8 pad: kills bank conflicts)
#define TILE_P    256                     // patch rows per workgroup
#define NWAVES    16                      // 512 threads; wave pair shares 32 rows, splits 256 words
#define LDS_BYTES (W_DIM * LDS_PITCH * 2) // 266,240 B — fits 320 KB WGP LDS

// pack two fp32 -> two bf16 in one dword: round-half-up (+0x8000) then one
// v_perm_b32 splicing the high halves -> 3 VALU per pair.
__device__ __forceinline__ unsigned int pkbf(float x, float y) {
#if __has_builtin(__builtin_amdgcn_cvt_pk_bf16_f32)
  union { v2bf v; unsigned int u; } cv;
  cv.v = __builtin_amdgcn_cvt_pk_bf16_f32(x, y);
  return cv.u;
#elif __has_builtin(__builtin_amdgcn_perm)
  unsigned int rx = __float_as_uint(x) + 0x8000u;
  unsigned int ry = __float_as_uint(y) + 0x8000u;
  return __builtin_amdgcn_perm(ry, rx, 0x07060302u);  // {ry[31:16], rx[31:16]}
#else
  unsigned int rx = (__float_as_uint(x) + 0x8000u) >> 16;
  unsigned int ry = (__float_as_uint(y) + 0x8000u) & 0xFFFF0000u;
  return ry | rx;
#endif
}

// order-preserving float<->uint map for atomicMax on floats (finite inputs)
__device__ __forceinline__ unsigned int fmap(float x) {
  unsigned int u = __float_as_uint(x);
  return (u & 0x80000000u) ? ~u : (u | 0x80000000u);
}
__device__ __forceinline__ float funmap(unsigned int m) {
  unsigned int u = (m & 0x80000000u) ? (m ^ 0x80000000u) : ~m;
  return __uint_as_float(u);
}

extern "C" __global__ void spwa_init(unsigned int* __restrict__ colmax_u,
                                     float* __restrict__ rowsum) {
  int i = blockIdx.x * 256 + threadIdx.x;
  if (i < B_BATCH * W_DIM) colmax_u[i] = 0u;   // 0 == fmap minimum (below fmap(-inf))
  if (i < B_BATCH)         rowsum[i]   = 0.0f;
}

extern "C" __global__ void __launch_bounds__(512, 1)
spwa_main(const float* __restrict__ patches, const float* __restrict__ text,
          unsigned int* __restrict__ colmax_u, float* __restrict__ rowsum) {
  extern __shared__ unsigned short smem[];
  const int tid  = threadIdx.x;
  const int lane = tid & 31;
  const int wv   = tid >> 5;
  const int pr   = wv >> 1;        // wave-pair index: 8 pairs x 32 rows = 256 rows
  const int h    = wv & 1;         // word half: 0 -> w 0..127, 1 -> w 128..255
  const int b    = blockIdx.y;
  const int p0   = blockIdx.x * TILE_P + pr * 32;

  // ---- Stage text[b] (256x512 fp32) into LDS as bf16, padded pitch ----
  const float* tb = text + (size_t)b * W_DIM * D_DIM;
  for (int idx = tid; idx < (W_DIM * D_DIM) / 4; idx += 512) {
    int w = idx >> 7;              // 128 float4 per row
    int d = (idx & 127) << 2;
    f32x4 v = *(const f32x4*)(tb + ((size_t)w << 9) + d);
    union { unsigned int u[2]; v4u pk; } cv;
    cv.u[0] = pkbf(v.x, v.y);
    cv.u[1] = pkbf(v.z, v.w);
    *(v4u*)(smem + w * LDS_PITCH + d) = cv.pk;
  }
  __syncthreads();

  const int grp = lane >> 4;       // 0: lanes 0-15, 1: lanes 16-31
  const int lm  = lane & 15;
  // RT loads: the other wave of the pair re-reads these rows -> hits WGP$/L2.
  const float* aRow0 = patches + ((size_t)(b * P_DIM + p0 + lm)) * D_DIM;
  const float* aRow1 = aRow0 + (size_t)16 * D_DIM;

  // 2 M-tiles x 8 N-tiles: 128 acc VGPRs -> stays under 256, no vgpr-msb churn
  v8f acc0[8], acc1[8];
  v8f z = {0.f, 0.f, 0.f, 0.f, 0.f, 0.f, 0.f, 0.f};
#pragma unroll
  for (int t = 0; t < 8; ++t) { acc0[t] = z; acc1[t] = z; }

  union Au { unsigned int ud[8];    v16bf b; };
  union Bu { v8u h[2];              v16bf b; };

#pragma unroll
  for (int kk = 0; kk < D_DIM / 32; ++kk) {
    const int kb = kk * 32;
    // A fragments (16x32 bf16, ISA layout), two M-tiles per wave:
    // VGPR v holds K = (v%4)*2 + (v/4)*16 + grp*8 -> merged b128 loads.
    Au a0, a1;
#pragma unroll
    for (int v = 0; v < 8; ++v) {
      const int koff = ((v & 3) << 1) + ((v >> 2) << 4) + (grp << 3);
      f32x2 v0 = *(const f32x2*)(aRow0 + kb + koff);
      f32x2 v1 = *(const f32x2*)(aRow1 + kb + koff);
      a0.ud[v] = pkbf(v0.x, v0.y);
      a1.ud[v] = pkbf(v1.x, v1.y);
    }
    // B fragments (32x16 bf16, ISA layout) for this wave's 8 word-tiles;
    // each LDS fragment feeds TWO WMMAs -> LDS bytes/FLOP halved vs 1 M-tile.
    const int dbase = kb + (grp << 3);
#pragma unroll
    for (int t = 0; t < 8; ++t) {
      Bu bf;
      const int w = h * 128 + t * 16 + lm;
      const v8u* bp = (const v8u*)(smem + w * LDS_PITCH + dbase);
      bf.h[0] = bp[0];
      bf.h[1] = bp[2];
      acc0[t] = __builtin_amdgcn_wmma_f32_16x16x32_bf16(
          false, a0.b, false, bf.b, (short)0, acc0[t], false, false);
      acc1[t] = __builtin_amdgcn_wmma_f32_16x16x32_bf16(
          false, a1.b, false, bf.b, (short)0, acc1[t], false, false);
    }
  }

  // ---- Fused reductions. C/D layout: VGPR i, lanes<16 -> (M=i, N=lane),
  //      lanes>=16 -> (M=i+8, N=lane-16). ----
  __syncthreads();                               // all waves done reading B; LDS reusable
  float* cw  = (float*)smem;                     // [8 pairs][256] col maxes
  float* rm  = (float*)smem + 8 * 256;           // [16 waves][32] partial row maxes
  float* rps = (float*)smem + 8 * 256 + 16 * 32; // [8] block row-sum partials

  // partial row maxes over this wave's 128 words (half of W) for its 32 rows
#pragma unroll
  for (int mt = 0; mt < 2; ++mt) {
#pragma unroll
    for (int i = 0; i < 8; ++i) {
      float m = mt ? acc1[0][i] : acc0[0][i];
#pragma unroll
      for (int t = 1; t < 8; ++t) m = fmaxf(m, mt ? acc1[t][i] : acc0[t][i]);
      m = fmaxf(m, __shfl_xor(m, 1, 32));
      m = fmaxf(m, __shfl_xor(m, 2, 32));
      m = fmaxf(m, __shfl_xor(m, 4, 32));
      m = fmaxf(m, __shfl_xor(m, 8, 32));
      if (lm == 0) rm[wv * 32 + mt * 16 + grp * 8 + i] = m;   // lanes 0 and 16 write
    }
  }
  // col maxes over this wave's 32 rows per word
#pragma unroll
  for (int t = 0; t < 8; ++t) {
    float m = fmaxf(acc0[t][0], acc1[t][0]);
#pragma unroll
    for (int i = 1; i < 8; ++i) m = fmaxf(m, fmaxf(acc0[t][i], acc1[t][i]));
    m = fmaxf(m, __shfl_xor(m, 16, 32));         // combine both M-halves
    if (grp == 0) cw[pr * 256 + h * 128 + t * 16 + lm] = m;
  }
  __syncthreads();

  if (tid < 256) {
    // word -> patch: combine the 8 row-pair blocks, one word per thread
    float m = cw[tid];
#pragma unroll
    for (int p = 1; p < 8; ++p) m = fmaxf(m, cw[p * 256 + tid]);
    atomicMax(&colmax_u[b * 256 + tid], fmap(m));

    // patch -> word: max across the wave pair's word-halves, then block-sum
    const int p = tid >> 5, r = tid & 31;        // 8 pairs x 32 rows
    float rmx = fmaxf(rm[(2 * p) * 32 + r], rm[(2 * p + 1) * 32 + r]);
    rmx += __shfl_xor(rmx, 1, 32);
    rmx += __shfl_xor(rmx, 2, 32);
    rmx += __shfl_xor(rmx, 4, 32);
    rmx += __shfl_xor(rmx, 8, 32);
    rmx += __shfl_xor(rmx, 16, 32);
    if (lane == 0) rps[wv] = rmx;                // waves 0..7 only (tid<256)
  }
  __syncthreads();
  if (tid == 0) {
    float s = 0.f;
#pragma unroll
    for (int p = 0; p < 8; ++p) s += rps[p];
    atomicAdd(&rowsum[b], s);
  }
}

extern "C" __global__ void spwa_finalize(const unsigned int* __restrict__ colmax_u,
                                         const float* __restrict__ rowsum,
                                         float* __restrict__ out) {
  __shared__ float ws[8];
  const int b = blockIdx.x, tid = threadIdx.x;
  float v = funmap(colmax_u[b * 256 + tid]);
  v += __shfl_xor(v, 1, 32);
  v += __shfl_xor(v, 2, 32);
  v += __shfl_xor(v, 4, 32);
  v += __shfl_xor(v, 8, 32);
  v += __shfl_xor(v, 16, 32);
  if ((tid & 31) == 0) ws[tid >> 5] = v;
  __syncthreads();
  if (tid == 0) {
    float s = 0.f;
#pragma unroll
    for (int i = 0; i < 8; ++i) s += ws[i];
    out[b] = 0.5f * (rowsum[b] * (1.0f / P_DIM) + s * (1.0f / W_DIM));
  }
}

extern "C" void kernel_launch(void* const* d_in, const int* in_sizes, int n_in,
                              void* d_out, int out_size, void* d_ws, size_t ws_size,
                              hipStream_t stream) {
  (void)in_sizes; (void)n_in; (void)out_size; (void)ws_size;
  const float* patches = (const float*)d_in[0];   // [64,4096,512] fp32
  const float* text    = (const float*)d_in[1];   // [64,256,512]  fp32
  float* out = (float*)d_out;                     // [64] fp32

  unsigned int* colmax_u = (unsigned int*)d_ws;                       // 64*256 u32
  float* rowsum = (float*)((char*)d_ws + B_BATCH * W_DIM * sizeof(unsigned int)); // 64 f32

  (void)hipFuncSetAttribute((const void*)spwa_main,
                            hipFuncAttributeMaxDynamicSharedMemorySize, LDS_BYTES);

  spwa_init<<<B_BATCH, 256, 0, stream>>>(colmax_u, rowsum);
  spwa_main<<<dim3(P_DIM / TILE_P, B_BATCH), 512, LDS_BYTES, stream>>>(
      patches, text, colmax_u, rowsum);
  spwa_finalize<<<B_BATCH, 256, 0, stream>>>(colmax_u, rowsum, out);
}
